// RoIHead_25323127177302
// MI455X (gfx1250) — matrix operand order repaired and checked by reading, not compiled, not run on hardware
//
#include <hip/hip_runtime.h>
#include <hip/hip_bf16.h>

// ---------------------------------------------------------------------------
// RoIHead for MI455X (gfx1250, wave32, WMMA).
//   k1: roi max-pool (fp32 math) -> pooled activations as bf16 [256, 12544]
//   k2: fc6 GEMM  (A bf16 [256,12544]) x (W fp32 [12544,4096] cvt->bf16 in-reg)
//       + bias + relu -> bf16 [256,4096]       (v_wmma_f32_16x16x32_bf16)
//   k3: fc7 GEMM  (A bf16 [256,4096])  x (W fp32 [4096,4096])
//       + bias + relu -> fp32 [256,4096]
//   k4: heads (4096 -> 84 loc, 4096 -> 21 cls), plain FMA (tiny)
// Weights are streamed from HBM exactly once (272 MB fp32 ~ 11.7us @ 23.3TB/s);
// activations stay L2-resident. f32->bf16 uses native hardware converts.
// ---------------------------------------------------------------------------

typedef __bf16 v16bf __attribute__((ext_vector_type(16)));
typedef float  v8f   __attribute__((ext_vector_type(8)));

#define POOL   7
#define NCH    256
#define NROI   256
#define DIN    (NCH * POOL * POOL)   // 12544
#define DHID   4096
#define NLOC   84
#define NCLS   21

union AFrag { uint4 q[2]; v16bf v; };

// ---------------------------------------------------------------------------
// Kernel 1: RoI max pool. grid = (NROI, 49), block = 256 (one thread/channel).
// ---------------------------------------------------------------------------
__global__ __launch_bounds__(256) void roi_pool_bf16(
    const float* __restrict__ feat3,   // [256,100,100]
    const float* __restrict__ feat4,   // [256, 50, 50]
    const float* __restrict__ feat5,   // [256, 25, 25]
    const float* __restrict__ rois,    // [256, 4] (y1,x1,y2,x2)
    __bf16* __restrict__ pooled)       // bf16 [256, 12544]
{
  const int n   = blockIdx.x;
  const int bin = blockIdx.y;          // ph*7+pw
  const int ph  = bin / POOL;
  const int pw  = bin % POOL;
  const int c   = threadIdx.x;

  const float y1 = rois[4 * n + 0], x1 = rois[4 * n + 1];
  const float y2 = rois[4 * n + 2], x2 = rois[4 * n + 3];

  // FPN level: clip(round(ln(sqrt(HW)/224)+5), 3, 5) - 3
  float lv = rintf(logf(sqrtf((y2 - y1) * (x2 - x1)) / 224.0f) + 5.0f);
  lv = fminf(fmaxf(lv, 3.0f), 5.0f);
  const int lvl = (int)lv - 3;

  const float scale = (lvl == 0) ? 0.125f : (lvl == 1) ? 0.0625f : 0.03125f;
  const int   HW    = (lvl == 0) ? 100 : (lvl == 1) ? 50 : 25;
  const float hwf   = (float)HW;
  const float* __restrict__ f =
      ((lvl == 0) ? feat3 : (lvl == 1) ? feat4 : feat5) + (size_t)c * HW * HW;

  const float ys = floorf(y1 * scale + 0.5f);
  const float xs = floorf(x1 * scale + 0.5f);
  const float ye = floorf(y2 * scale + 0.5f);
  const float xe = floorf(x2 * scale + 0.5f);
  const float bh = fmaxf(ye - ys + 1.0f, 1.0f) / (float)POOL;
  const float bw = fmaxf(xe - xs + 1.0f, 1.0f) / (float)POOL;

  const int hs = (int)fminf(fmaxf(floorf((float)ph * bh) + ys, 0.0f), hwf);
  const int he = (int)fminf(fmaxf(ceilf((float)(ph + 1) * bh) + ys, 0.0f), hwf);
  const int ws = (int)fminf(fmaxf(floorf((float)pw * bw) + xs, 0.0f), hwf);
  const int we = (int)fminf(fmaxf(ceilf((float)(pw + 1) * bw) + xs, 0.0f), hwf);

  float m = -__builtin_inff();
  for (int r = hs; r < he; ++r) {
    const float* row = f + (size_t)r * HW;
    for (int q = ws; q < we; ++q) m = fmaxf(m, row[q]);
  }
  const float outv = (he <= hs || we <= ws) ? 0.0f : m;

  pooled[(size_t)n * DIN + (size_t)c * (POOL * POOL) + bin] = (__bf16)outv;
}

// ---------------------------------------------------------------------------
// Kernel 2/3: WMMA GEMM.  out[M=256, N] = relu(A[256,K] * B[K,N] + bias)
//   A: bf16 row-major (L2-resident).  B: fp32 row-major, cvt->bf16 in-register
//   (native v_cvt).  block = 256 thr = 8 waves; WG covers 256 rows x 64 cols.
//   wave w: col tile (w&3), row tiles ( (w>>2)*128 + i*16 ), i=0..7.
// ---------------------------------------------------------------------------
template <bool OUT_BF16>
__global__ __launch_bounds__(256) void gemm_wmma_bf16(
    const __bf16* __restrict__ A,     // bf16 [256, K]
    const float*  __restrict__ B,     // fp32 [K, N]
    const float*  __restrict__ bias,  // fp32 [N]
    void*         __restrict__ out,   // [256, N] bf16 or fp32
    int K, int N)
{
  const int lane    = threadIdx.x & 31;
  const int wave    = threadIdx.x >> 5;
  const int ct      = wave & 3;
  const int rh      = wave >> 2;
  const int nbase   = blockIdx.x * 64 + ct * 16;
  const int n       = nbase + (lane & 15);      // this lane's output column
  const int rowbase = rh * 128;
  const int kbA     = (lane >> 4) * 8;          // A-fragment K sub-base
  const int kbB     = (lane >> 4) * 8;          // B-fragment K sub-base
  const int rA      = lane & 15;                // A-fragment row within tile

  const v8f zero = {0.f, 0.f, 0.f, 0.f, 0.f, 0.f, 0.f, 0.f};
  v8f acc[8];
#pragma unroll
  for (int i = 0; i < 8; ++i) acc[i] = zero;

  for (int k0 = 0; k0 < K; k0 += 32) {
    // ---- B fragment: 16 column-strided fp32 loads, native cvt -> bf16 ----
    const float* bp = B + (size_t)(k0 + kbB) * N + n;
    v16bf bfrag;
#pragma unroll
    for (int j = 0; j < 16; ++j) bfrag[j] = (__bf16)bp[(size_t)(j + (j & 8)) * N];

    // prefetch next weight K-slab (speculative global_prefetch_b8)
    __builtin_prefetch(bp + (size_t)32 * N, 0, 1);

    // ---- 8 row tiles: A fragment (2 x 16B vector loads) + WMMA ----
#pragma unroll
    for (int i = 0; i < 8; ++i) {
      const __bf16* ap = A + (size_t)(rowbase + i * 16 + rA) * K + k0 + kbA;
      AFrag af;
      af.q[0] = *(const uint4*)(ap);        // K = kbA + 0..7
      af.q[1] = *(const uint4*)(ap + 16);   // K = kbA + 16..23
      acc[i] = __builtin_amdgcn_wmma_f32_16x16x32_bf16(
          false, af.v, false, bfrag, (short)0, acc[i], false, false);
    }
  }

  // ---- epilogue: bias + relu, store per C/D VGPR layout ----
  const float bv = bias[n];
  float*  outf = (float*)out;
  __bf16* outh = (__bf16*)out;
#pragma unroll
  for (int i = 0; i < 8; ++i) {
    const int mbase = rowbase + i * 16 + 8 * (lane >> 4);
#pragma unroll
    for (int v = 0; v < 8; ++v) {
      float x = fmaxf(acc[i][v] + bv, 0.0f);
      const size_t off = (size_t)(mbase + v) * N + n;
      if (OUT_BF16) outh[off] = (__bf16)x;
      else          outf[off] = x;
    }
  }
}

// ---------------------------------------------------------------------------
// Kernel 4: tiny heads. grid = NROI, block = 128 (105 active threads).
//   d_out = [ roi_cls_locs 256x84 | roi_scores 256x21 ] (fp32, flat).
// ---------------------------------------------------------------------------
__global__ __launch_bounds__(128) void roi_heads(
    const float* __restrict__ fc7,   // [256, 4096]
    const float* __restrict__ Wloc,  // [4096, 84]
    const float* __restrict__ bloc,  // [84]
    const float* __restrict__ Wcls,  // [4096, 21]
    const float* __restrict__ bcls,  // [21]
    float* __restrict__ outp)
{
  const int roi = blockIdx.x;
  const int t   = threadIdx.x;
  if (t >= NLOC + NCLS) return;

  const float* __restrict__ x = fc7 + (size_t)roi * DHID;

  const bool   is_loc = (t < NLOC);
  const int    col    = is_loc ? t : (t - NLOC);
  const int    ldw    = is_loc ? NLOC : NCLS;
  const float* __restrict__ w = (is_loc ? Wloc : Wcls) + col;
  float s = is_loc ? bloc[col] : bcls[col];

  for (int k = 0; k < DHID; ++k) s = fmaf(x[k], w[(size_t)k * ldw], s);

  if (is_loc) outp[(size_t)roi * NLOC + col] = s;
  else        outp[(size_t)NROI * NLOC + (size_t)roi * NCLS + col] = s;
}

// ---------------------------------------------------------------------------
extern "C" void kernel_launch(void* const* d_in, const int* in_sizes, int n_in,
                              void* d_out, int out_size, void* d_ws, size_t ws_size,
                              hipStream_t stream) {
  const float* feat3 = (const float*)d_in[0];
  const float* feat4 = (const float*)d_in[1];
  const float* feat5 = (const float*)d_in[2];
  const float* rois  = (const float*)d_in[3];
  const float* Wfc6  = (const float*)d_in[4];
  const float* bfc6  = (const float*)d_in[5];
  const float* Wfc7  = (const float*)d_in[6];
  const float* bfc7  = (const float*)d_in[7];
  const float* Wloc  = (const float*)d_in[8];
  const float* bloc  = (const float*)d_in[9];
  const float* Wcls  = (const float*)d_in[10];
  const float* bcls  = (const float*)d_in[11];

  // workspace: pooled bf16 | fc6 bf16 | fc7 fp32   (~12.7 MB total)
  char* ws = (char*)d_ws;
  size_t off = 0;
  __bf16* pooled = (__bf16*)(ws + off);
  off += ((size_t)NROI * DIN * sizeof(__bf16) + 255) & ~(size_t)255;
  __bf16* fc6o   = (__bf16*)(ws + off);
  off += ((size_t)NROI * DHID * sizeof(__bf16) + 255) & ~(size_t)255;
  float*  fc7o   = (float*)(ws + off);

  // 1) RoI max pool -> bf16 activations
  roi_pool_bf16<<<dim3(NROI, POOL * POOL), 256, 0, stream>>>(
      feat3, feat4, feat5, rois, pooled);

  // 2) fc6: [256,12544] x [12544,4096] -> relu -> bf16
  gemm_wmma_bf16<true><<<DHID / 64, 256, 0, stream>>>(
      pooled, Wfc6, bfc6, (void*)fc6o, DIN, DHID);

  // 3) fc7: [256,4096] x [4096,4096] -> relu -> fp32
  gemm_wmma_bf16<false><<<DHID / 64, 256, 0, stream>>>(
      fc6o, Wfc7, bfc7, (void*)fc7o, DHID, DHID);

  // 4) heads
  roi_heads<<<NROI, 128, 0, stream>>>(fc7o, Wloc, bloc, Wcls, bcls,
                                      (float*)d_out);
}
